// CustomMultiHeadAttention_58445914964313
// MI455X (gfx1250) — matrix operand order repaired and checked by reading, not compiled, not run on hardware
//
#include <hip/hip_runtime.h>

// ---------------------------------------------------------------------------
// Causal MHA forward for MI455X (gfx1250, wave32, WMMA + Tensor Data Mover).
// B=512, T=64, C=768, H=12, D=64.
// Bandwidth-bound (~161 GFLOP vs ~700 MB HBM traffic @ 23.3 TB/s), so:
//   * x pre-converted to bf16 once; all intermediates (Q,K,Vt,P,O) bf16
//   * all GEMMs use v_wmma_f32_16x16x32_bf16 (fp32 accumulate)
//   * all LDS tile loads via TDM (tensor_load_to_lds, TENSORcnt),
//     double-buffered in the GEMM k-loops to overlap DMA with WMMA
// Workspace: ~256 MB of d_ws (bf16 x, weights, Q/K/Vt/Ocat).
// ---------------------------------------------------------------------------

typedef __bf16 bf16_t;
typedef __attribute__((ext_vector_type(16))) __bf16        v16bf;
typedef __attribute__((ext_vector_type(8)))  float         v8f;
typedef __attribute__((ext_vector_type(4)))  unsigned int  v4u;
typedef __attribute__((ext_vector_type(8)))  int           v8i;
typedef __attribute__((ext_vector_type(4)))  int           v4i;

#define NB  512   // batch
#define NT  64    // sequence length
#define NC  768   // channels
#define NH  12    // heads
#define ND  64    // head dim
#define WELEMS (NH * ND * NC)          // 589824 elements per weight tensor

union FragBF { v16bf v; uint4 q[2]; };

// A-matrix fragment (16x32 bf16, MxK), tile row-major [16][ldk] in LDS.
__device__ inline v16bf ld_frag_a(const bf16_t* tile, int ldk, int lane) {
  FragBF f;
  const bf16_t* p = tile + (lane & 15) * ldk + ((lane >> 4) << 3);
  f.q[0] = *reinterpret_cast<const uint4*>(p);
  f.q[1] = *reinterpret_cast<const uint4*>(p + 16);
  return f.v;
}

// B-matrix fragment (32x16 bf16, KxN) supplied as B^T row-major: tile[n][k].
__device__ inline v16bf ld_frag_b(const bf16_t* tile, int ldk, int lane) {
  FragBF f;
  const bf16_t* p = tile + (lane & 15) * ldk + ((lane >> 4) << 3);
  f.q[0] = *reinterpret_cast<const uint4*>(p);
  f.q[1] = *reinterpret_cast<const uint4*>(p + 16);
  return f.v;
}

// C/D layout: VGPR r of lane L holds element [M = r + 8*(L>>4)][N = L&15].
__device__ inline void st_cd_bf16(bf16_t* out, int ldn, const v8f& acc, int lane) {
  int n = lane & 15, hi = lane >> 4;
#pragma unroll
  for (int r = 0; r < 8; ++r) out[(r + 8 * hi) * ldn + n] = (bf16_t)acc[r];
}

__device__ inline void st_cd_bf16_T(bf16_t* out, int ldn, const v8f& acc, int lane) {
  int n = lane & 15, hi = lane >> 4;
#pragma unroll
  for (int r = 0; r < 8; ++r) out[n * ldn + (r + 8 * hi)] = (bf16_t)acc[r];
}

__device__ inline void st_cd_f32s(float* out, int ldn, const v8f& acc, float scale, int lane) {
  int n = lane & 15, hi = lane >> 4;
#pragma unroll
  for (int r = 0; r < 8; ++r) out[(r + 8 * hi) * ldn + n] = acc[r] * scale;
}

__device__ inline v8f wmma_bf16(const v16bf& a, const v16bf& b, const v8f& c) {
  return __builtin_amdgcn_wmma_f32_16x16x32_bf16(false, a, false, b, (short)0, c,
                                                 false, false);
}

// ---------------------------------------------------------------------------
// Tensor Data Mover: DMA a 2-D bf16 tile [tile_h rows x tile_w elems] with
// row stride `stride_elems` from global into LDS (rows packed contiguously).
// D# per CDNA5 ISA: group0 = {count/lds_addr/global_addr/
// type=2}, group1 = {data_size=1(2B), tensor dims, tile dims, dim0 stride}.
// Groups 2/3 unused (<=2D tensor). Issue from ONE wave only.
// ---------------------------------------------------------------------------
__device__ __forceinline__ void tdm_load_2d_bf16(void* lds_dst, const bf16_t* gsrc,
                                                 unsigned tile_w, unsigned tile_h,
                                                 unsigned stride_elems) {
  // LDS aperture maps generic addr[31:0] -> LDS byte offset.
  unsigned lds_byte = (unsigned)(unsigned long long)lds_dst;
  unsigned long long ga = (unsigned long long)gsrc;
  v4u g0;
  g0.x = 1u;                                               // count=1 (valid)
  g0.y = lds_byte;                                         // lds_addr
  g0.z = (unsigned)ga;                                     // global_addr[31:0]
  g0.w = ((unsigned)(ga >> 32) & 0x01FFFFFFu) | 0x80000000u; // addr[56:32] | type=2
  v8i g1;
  g1[0] = 0x00010000;                                      // mask=0, data_size=1 (2B)
  g1[1] = (int)((stride_elems & 0xFFFFu) << 16);           // tensor_dim0[15:0]
  g1[2] = (int)(((stride_elems >> 16) & 0xFFFFu)           // tensor_dim0[31:16]
                | ((tile_h & 0xFFFFu) << 16));             // tensor_dim1[15:0]
  g1[3] = (int)((tile_w & 0xFFFFu) << 16);                 // tile_dim0
  g1[4] = (int)(tile_h & 0xFFFFu);                         // tile_dim1 (tile_dim2=0)
  g1[5] = (int)stride_elems;                               // tensor_dim0_stride[31:0]
  g1[6] = 0;
  g1[7] = 0;
  v4i z4 = {0, 0, 0, 0};
#if defined(__clang_major__) && (__clang_major__ >= 23)
  v8i z8 = {0, 0, 0, 0, 0, 0, 0, 0};
  __builtin_amdgcn_tensor_load_to_lds(g0, g1, z4, z4, z8, 0);
#else
  __builtin_amdgcn_tensor_load_to_lds(g0, g1, z4, z4, 0);
#endif
}

// ---------------------------------------------------------------------------
// Stage 1: fp32 -> bf16 conversions (x and weights)
// ---------------------------------------------------------------------------
__global__ __launch_bounds__(256) void cvt_f32_to_bf16(const float* __restrict__ src,
                                                       bf16_t* __restrict__ dst, int n) {
  int i = blockIdx.x * blockDim.x + threadIdx.x;
  if (i < n) dst[i] = (bf16_t)src[i];
}

// ---------------------------------------------------------------------------
// Stage 2: QKV projection. grid = (B, H, 3[q,k,v]); block = 256 (8 waves).
// out[t][d] = sum_c xb[b,t,c]*w[z,h,d,c]; V stored transposed [D][T].
// Double-buffered TDM tile loads.
// ---------------------------------------------------------------------------
__global__ __launch_bounds__(256) void qkv_kernel(const bf16_t* __restrict__ xb,
                                                  const bf16_t* __restrict__ wqkv,
                                                  bf16_t* __restrict__ Qg,
                                                  bf16_t* __restrict__ Kg,
                                                  bf16_t* __restrict__ Vtg) {
  const int b = blockIdx.x, h = blockIdx.y, z = blockIdx.z;
  const int tid = threadIdx.x, lane = tid & 31, wave = tid >> 5;
  const int mt = wave >> 1, nt0 = (wave & 1) * 2;

  __shared__ alignas(16) bf16_t lds[8192];   // 2 x (A[64][32] + B[64][32]) = 16 KB

  const bf16_t* xrow = xb + (size_t)b * NT * NC;                   // [64][768]
  const bf16_t* wrow = wqkv + ((size_t)(z * NH + h) * ND) * NC;    // [64][768]

  if (wave == 0) {  // prologue: chunk 0 into buffer 0
    tdm_load_2d_bf16(lds,        xrow, 32, 64, NC);
    tdm_load_2d_bf16(lds + 2048, wrow, 32, 64, NC);
  }

  v8f acc0 = {}, acc1 = {};
  for (int kc = 0; kc < NC / 32; ++kc) {
    const int cur = kc & 1;
    bf16_t* As = lds + (cur ? 4096 : 0);
    bf16_t* Bs = As + 2048;
    if (wave == 0) {
      if (kc + 1 < NC / 32) {   // prefetch next chunk into the other buffer
        bf16_t* An = lds + (cur ? 0 : 4096);
        tdm_load_2d_bf16(An,        xrow + (kc + 1) * 32, 32, 64, NC);
        tdm_load_2d_bf16(An + 2048, wrow + (kc + 1) * 32, 32, 64, NC);
        __builtin_amdgcn_s_wait_tensorcnt(2);   // current chunk complete
      } else {
        __builtin_amdgcn_s_wait_tensorcnt(0);
      }
    }
    __syncthreads();            // publish DMA'd tiles to all waves
    v16bf a  = ld_frag_a(As + mt * 16 * 32, 32, lane);
    v16bf b0 = ld_frag_b(Bs + nt0 * 16 * 32, 32, lane);
    v16bf b1 = ld_frag_b(Bs + (nt0 + 1) * 16 * 32, 32, lane);
    acc0 = wmma_bf16(a, b0, acc0);
    acc1 = wmma_bf16(a, b1, acc1);
    __syncthreads();            // done reading before buffer is re-filled
  }

  bf16_t* ob = lds;             // [64][64] output staging (8 KB)
  if (z == 2) {
    st_cd_bf16_T(ob + (nt0 * 16) * 64 + mt * 16, 64, acc0, lane);
    st_cd_bf16_T(ob + ((nt0 + 1) * 16) * 64 + mt * 16, 64, acc1, lane);
  } else {
    st_cd_bf16(ob + (mt * 16) * 64 + nt0 * 16, 64, acc0, lane);
    st_cd_bf16(ob + (mt * 16) * 64 + (nt0 + 1) * 16, 64, acc1, lane);
  }
  __syncthreads();
  bf16_t* dst = (z == 0 ? Qg : z == 1 ? Kg : Vtg) + (((size_t)b * NH + h) << 12);
  {   // 4096 bf16, 16 per thread, coalesced b128 stores
    int flat = tid * 16;
    *reinterpret_cast<uint4*>(dst + flat)     = *reinterpret_cast<const uint4*>(ob + flat);
    *reinterpret_cast<uint4*>(dst + flat + 8) = *reinterpret_cast<const uint4*>(ob + flat + 8);
  }
}

// ---------------------------------------------------------------------------
// Stage 3: fused attention per (b,h). grid = (B, H); block = 256.
// Q,K,Vt DMA'd to LDS via TDM; S=scale*QK^T -> causal softmax -> O=PV.
// ---------------------------------------------------------------------------
__global__ __launch_bounds__(256) void attn_kernel(const bf16_t* __restrict__ Qg,
                                                   const bf16_t* __restrict__ Kg,
                                                   const bf16_t* __restrict__ Vtg,
                                                   bf16_t* __restrict__ Ocat) {
  const int b = blockIdx.x, h = blockIdx.y;
  const int tid = threadIdx.x, lane = tid & 31, wave = tid >> 5;
  const int mt = wave >> 1, nt0 = (wave & 1) * 2;

  __shared__ alignas(16) bf16_t qs[64 * 64];   // Q, then reused as P
  __shared__ alignas(16) bf16_t ks[64 * 64];   // K, then reused as O staging
  __shared__ alignas(16) bf16_t vs[64 * 64];   // V^T [d][s]
  __shared__ alignas(16) float  ss[64 * 64];   // S scores (fp32)

  const size_t base = ((size_t)b * NH + h) << 12;  // *4096
  if (wave == 0) {   // TDM block loads of Q, K, Vt (contiguous 64x64 tiles)
    tdm_load_2d_bf16(qs, Qg + base, 64, 64, 64);
    tdm_load_2d_bf16(ks, Kg + base, 64, 64, 64);
    tdm_load_2d_bf16(vs, Vtg + base, 64, 64, 64);
    __builtin_amdgcn_s_wait_tensorcnt(0);
  }
  __syncthreads();

  // S = Q K^T : A=Q[t][d], B[k=d][n=s]=K[s][d] (ks row-major [s][d])
  v8f s0 = {}, s1 = {};
#pragma unroll
  for (int k0 = 0; k0 < ND; k0 += 32) {
    v16bf a  = ld_frag_a(qs + mt * 16 * 64 + k0, 64, lane);
    v16bf b0 = ld_frag_b(ks + nt0 * 16 * 64 + k0, 64, lane);
    v16bf b1 = ld_frag_b(ks + (nt0 + 1) * 16 * 64 + k0, 64, lane);
    s0 = wmma_bf16(a, b0, s0);
    s1 = wmma_bf16(a, b1, s1);
  }
  const float scale = 0.125f;  // D^-0.5 = 1/8
  st_cd_f32s(ss + (mt * 16) * 64 + nt0 * 16, 64, s0, scale, lane);
  st_cd_f32s(ss + (mt * 16) * 64 + (nt0 + 1) * 16, 64, s1, scale, lane);
  __syncthreads();

  // causal softmax, one row per thread (threads 0..63); P -> qs (bf16)
  if (tid < NT) {
    const int t = tid;
    float* row = ss + t * 64;
    float m = row[0];
    for (int s = 1; s <= t; ++s) m = fmaxf(m, row[s]);
    float sum = 0.f;
    for (int s = 0; s <= t; ++s) { float e = __expf(row[s] - m); sum += e; row[s] = e; }
    float inv = 1.f / sum;
    bf16_t* prow = qs + t * 64;
    for (int s = 0; s <= t; ++s) prow[s] = (bf16_t)(row[s] * inv);
    for (int s = t + 1; s < NT; ++s) prow[s] = (bf16_t)0.f;
  }
  __syncthreads();

  // O = P V : A=P[t][s], B[k=s][n=d]=V[s][d]=Vt[d][s] (vs row-major [d][s])
  v8f o0 = {}, o1 = {};
#pragma unroll
  for (int k0 = 0; k0 < NT; k0 += 32) {
    v16bf a  = ld_frag_a(qs + mt * 16 * 64 + k0, 64, lane);
    v16bf b0 = ld_frag_b(vs + nt0 * 16 * 64 + k0, 64, lane);
    v16bf b1 = ld_frag_b(vs + (nt0 + 1) * 16 * 64 + k0, 64, lane);
    o0 = wmma_bf16(a, b0, o0);
    o1 = wmma_bf16(a, b1, o1);
  }
  __syncthreads();
  bf16_t* ob = ks;  // reuse K staging for O [t][d]
  st_cd_bf16(ob + (mt * 16) * 64 + nt0 * 16, 64, o0, lane);
  st_cd_bf16(ob + (mt * 16) * 64 + (nt0 + 1) * 16, 64, o1, lane);
  __syncthreads();
  {   // write Ocat[b][t][h*64 + d], 16 elems/thread, 32B coalesced
    int t = tid >> 2, seg = (tid & 3) * 16;
    bf16_t* dst = Ocat + ((size_t)b * NT + t) * (NH * ND) + h * ND + seg;
    const bf16_t* src = ob + t * 64 + seg;
    *reinterpret_cast<uint4*>(dst)     = *reinterpret_cast<const uint4*>(src);
    *reinterpret_cast<uint4*>(dst + 8) = *reinterpret_cast<const uint4*>(src + 8);
  }
}

// ---------------------------------------------------------------------------
// Stage 4: output projection. out[m][n] = sum_k Ocat[m][k]*w_proj[n][k] + b[n]
// grid = (M/64=512, N/64=12); block = 256. Double-buffered TDM tile loads.
// ---------------------------------------------------------------------------
__global__ __launch_bounds__(256) void proj_kernel(const bf16_t* __restrict__ Ocat,
                                                   const bf16_t* __restrict__ wpb,
                                                   const float* __restrict__ bias,
                                                   float* __restrict__ out) {
  const int mblk = blockIdx.x, nblk = blockIdx.y;
  const int tid = threadIdx.x, lane = tid & 31, wave = tid >> 5;
  const int mt = wave >> 1, nt0 = (wave & 1) * 2;

  __shared__ alignas(16) bf16_t lds[8192];    // 2 x (A + B) tiles, 16 KB
  __shared__ alignas(16) float  of[64 * 64];  // fp32 output staging, 16 KB

  const bf16_t* abase = Ocat + (size_t)mblk * 64 * NC;  // [64 rows m][768 k]
  const bf16_t* bbase = wpb  + (size_t)nblk * 64 * NC;  // [64 rows n][768 k]

  if (wave == 0) {
    tdm_load_2d_bf16(lds,        abase, 32, 64, NC);
    tdm_load_2d_bf16(lds + 2048, bbase, 32, 64, NC);
  }

  v8f acc0 = {}, acc1 = {};
  for (int kc = 0; kc < NC / 32; ++kc) {
    const int cur = kc & 1;
    bf16_t* As = lds + (cur ? 4096 : 0);
    bf16_t* Bs = As + 2048;
    if (wave == 0) {
      if (kc + 1 < NC / 32) {
        bf16_t* An = lds + (cur ? 0 : 4096);
        tdm_load_2d_bf16(An,        abase + (kc + 1) * 32, 32, 64, NC);
        tdm_load_2d_bf16(An + 2048, bbase + (kc + 1) * 32, 32, 64, NC);
        __builtin_amdgcn_s_wait_tensorcnt(2);
      } else {
        __builtin_amdgcn_s_wait_tensorcnt(0);
      }
    }
    __syncthreads();
    v16bf a  = ld_frag_a(As + mt * 16 * 32, 32, lane);
    v16bf b0 = ld_frag_b(Bs + nt0 * 16 * 32, 32, lane);
    v16bf b1 = ld_frag_b(Bs + (nt0 + 1) * 16 * 32, 32, lane);
    acc0 = wmma_bf16(a, b0, acc0);
    acc1 = wmma_bf16(a, b1, acc1);
    __syncthreads();
  }
  st_cd_f32s(of + (mt * 16) * 64 + nt0 * 16, 64, acc0, 1.0f, lane);
  st_cd_f32s(of + (mt * 16) * 64 + (nt0 + 1) * 16, 64, acc1, 1.0f, lane);
  __syncthreads();
  {   // bias + fp32 store, 16 elems/thread
    int row = tid >> 2, seg = (tid & 3) * 16;
    const float4* s4 = reinterpret_cast<const float4*>(of + row * 64 + seg);
    const float4* b4 = reinterpret_cast<const float4*>(bias + nblk * 64 + seg);
    float4* d4 = reinterpret_cast<float4*>(
        out + ((size_t)(mblk * 64 + row)) * NC + nblk * 64 + seg);
#pragma unroll
    for (int i = 0; i < 4; ++i) {
      float4 v = s4[i], bb = b4[i];
      v.x += bb.x; v.y += bb.y; v.z += bb.z; v.w += bb.w;
      d4[i] = v;
    }
  }
}

// ---------------------------------------------------------------------------
extern "C" void kernel_launch(void* const* d_in, const int* in_sizes, int n_in,
                              void* d_out, int out_size, void* d_ws, size_t ws_size,
                              hipStream_t stream) {
  const float* x      = (const float*)d_in[0];
  const float* wq     = (const float*)d_in[1];
  const float* wk     = (const float*)d_in[2];
  const float* wv     = (const float*)d_in[3];
  const float* w_proj = (const float*)d_in[4];
  const float* b_proj = (const float*)d_in[5];
  float* out = (float*)d_out;

  // Workspace layout (bf16 elements); total ~256 MB.
  const size_t QKV_ELEMS = (size_t)NB * NH * NT * ND;   // 25165824
  const size_t X_ELEMS   = (size_t)NB * NT * NC;        // 25165824
  bf16_t* ws   = (bf16_t*)d_ws;
  bf16_t* wqkv = ws;                                  // [3][H][D][C]
  bf16_t* wpb  = wqkv + (size_t)3 * WELEMS;           // [C][H*D]
  bf16_t* xbuf = wpb + WELEMS;                        // [B][T][C] bf16
  bf16_t* Qg   = xbuf + X_ELEMS;                      // [B][H][T][D]
  bf16_t* Kg   = Qg + QKV_ELEMS;                      // [B][H][T][D]
  bf16_t* Vtg  = Kg + QKV_ELEMS;                      // [B][H][D][T]
  bf16_t* Ocat = Vtg + QKV_ELEMS;                     // [B][T][H*D]

  const int cvtBlocks = (WELEMS + 255) / 256;
  cvt_f32_to_bf16<<<cvtBlocks, 256, 0, stream>>>(wq, wqkv, WELEMS);
  cvt_f32_to_bf16<<<cvtBlocks, 256, 0, stream>>>(wk, wqkv + WELEMS, WELEMS);
  cvt_f32_to_bf16<<<cvtBlocks, 256, 0, stream>>>(wv, wqkv + 2 * (size_t)WELEMS, WELEMS);
  cvt_f32_to_bf16<<<cvtBlocks, 256, 0, stream>>>(w_proj, wpb, NC * NH * ND);
  cvt_f32_to_bf16<<<(int)((X_ELEMS + 255) / 256), 256, 0, stream>>>(x, xbuf, (int)X_ELEMS);

  qkv_kernel<<<dim3(NB, NH, 3), 256, 0, stream>>>(xbuf, wqkv, Qg, Kg, Vtg);
  attn_kernel<<<dim3(NB, NH), 256, 0, stream>>>(Qg, Kg, Vtg, Ocat);
  proj_kernel<<<dim3(NB * NT / 64, NC / 64), 256, 0, stream>>>(Ocat, wpb, b_proj, out);
}
